// RPN_2267742732673
// MI455X (gfx1250) — matrix operand order repaired and checked by reading, not compiled, not run on hardware
//
#include <hip/hip_runtime.h>

// ---------------- constants ----------------
#define NIMG   2
#define TOTP   87296          // sum of H*W over 5 levels, per image
#define TOTP3  261888         // 3 * TOTP (anchors per image)
#define KCOMB  4768           // 1000*4 + 768 combined candidates per image
#define NWORDS 149            // KCOMB / 32
#define IMGSZ  1024.0f
#define NMS_THR 0.7f
#define CLIPV  4.135166556742356f   // log(1000/16)

#if __has_builtin(__builtin_amdgcn_global_load_async_to_lds_b128) && \
    __has_builtin(__builtin_amdgcn_s_wait_asynccnt)
#define USE_ASYNC_LDS 1
#endif

typedef int v4i_vs __attribute__((vector_size(16)));   // matches builtin param pointee
typedef __attribute__((address_space(1))) v4i_vs gv4i;
typedef __attribute__((address_space(3))) v4i_vs lv4i;

__constant__ int   d_P[5]     = {65536, 16384, 4096, 1024, 256};
__constant__ int   d_prefP[5] = {0, 65536, 81920, 86016, 87040};
__constant__ int   d_off3[5]  = {0, 196608, 245760, 258048, 261120};
__constant__ int   d_dim[5]   = {256, 128, 64, 32, 16};
__constant__ float d_stride[5]= {4.f, 8.f, 16.f, 32.f, 64.f};
__constant__ float d_size[5]  = {32.f, 64.f, 128.f, 256.f, 512.f};
__constant__ int   d_ksel[5]  = {1000, 1000, 1000, 1000, 768};
__constant__ int   d_cbase[5] = {0, 1000, 2000, 3000, 4000};
__constant__ float d_ratio[3] = {0.5f, 1.0f, 2.0f};

// ---------------- helpers ----------------
__device__ __forceinline__ unsigned short f2bf(float f) {  // fp32 -> bf16 RNE
  unsigned u = __float_as_uint(f);
  u += 0x7FFFu + ((u >> 16) & 1u);
  return (unsigned short)(u >> 16);
}
__device__ __forceinline__ float bf2f(unsigned short h) {
  return __uint_as_float(((unsigned)h) << 16);
}
__device__ __forceinline__ unsigned fkey(float f) {        // monotonic float->uint
  unsigned u = __float_as_uint(f);
  return (u & 0x80000000u) ? ~u : (u | 0x80000000u);
}
__device__ __forceinline__ float neg_inf() { return __uint_as_float(0xFF800000u); }

typedef __attribute__((ext_vector_type(16))) __bf16 v16bf;
typedef __attribute__((ext_vector_type(8)))  float  v8f;

// ---------------- 0: zero atomic counters ----------------
__global__ void k_init(unsigned* cnt) {
  if (threadIdx.x < 64) cnt[threadIdx.x] = 0u;
}

// ---------------- 1: weights fp32 HWIO -> bf16 co-major [co][2304] ----------------
__global__ void k_wconv(const float* __restrict__ w, unsigned short* __restrict__ WT) {
  int t = blockIdx.x * 256 + threadIdx.x;
  if (t >= 2304 * 256) return;
  int co = t / 2304;
  int k  = t - co * 2304;
  WT[t] = f2bf(w[(size_t)k * 256 + co]);   // w is [k][co] flat
}

// ---------------- 2: 3x3 conv + ReLU via bf16 WMMA (implicit GEMM) ----------------
// one block: 2 rows x 16 cols = 32 positions x 256 output channels, K = 9*256 = 2304
__global__ void __launch_bounds__(256) k_conv3(const float* __restrict__ feat,
                                               const unsigned short* __restrict__ WT,
                                               const float* __restrict__ bconv,
                                               unsigned short* __restrict__ T,
                                               int H, int W, int posOff) {
  __shared__ __align__(16) unsigned short patch[4 * 18 * 256];  // 36864 B (rows y0-1..y0+2)
  __shared__ __align__(16) unsigned short wslab[256 * 32];      // 16384 B, [co][klocal]

  const int tid  = threadIdx.x;
  const int lane = tid & 31;
  const int wid  = tid >> 5;
  const int n    = blockIdx.y;
  const int tpr  = W >> 4;                 // 16-wide tiles per row
  const int rp   = blockIdx.x / tpr;       // row-pair index
  const int y0   = rp * 2;
  const int x0   = (blockIdx.x - rp * tpr) << 4;

  // stage zero-padded 4 x 18 x 256 input patch (fp32 -> bf16)
  for (int e = tid; e < 4 * 18 * 256; e += 256) {
    int ci  = e & 255;
    int col = (e >> 8) % 18;
    int row = (e >> 8) / 18;
    int gy = y0 + row - 1, gx = x0 + col - 1;
    float v = 0.0f;
    if (gy >= 0 && gy < H && gx >= 0 && gx < W)
      v = feat[(((size_t)n * H + gy) * W + gx) * 256 + ci];
    patch[e] = f2bf(v);
  }

  v8f acc00 = {0.f,0.f,0.f,0.f,0.f,0.f,0.f,0.f};  // row y0,   channels co0 tile
  v8f acc01 = {0.f,0.f,0.f,0.f,0.f,0.f,0.f,0.f};  // row y0,   channels co1 tile
  v8f acc10 = {0.f,0.f,0.f,0.f,0.f,0.f,0.f,0.f};  // row y0+1, channels co0 tile
  v8f acc11 = {0.f,0.f,0.f,0.f,0.f,0.f,0.f,0.f};  // row y0+1, channels co1 tile

  const int m   = lane & 15;                 // A-matrix row (x position within tile)
  const int kb  = (lane < 16) ? 0 : 8;       // A K-base per documented layout
  const int khU = (lane < 16) ? 0 : 4;       // B K-base (uint units: K=8 -> 4 uints)
  const int co0 = wid * 32 + (lane & 15);    // B/N column, subtile 0
  const int co1 = co0 + 16;                  // subtile 1

  __syncthreads();

  const unsigned* patchU = (const unsigned*)patch;
  const unsigned* wslabU = (const unsigned*)wslab;

  for (int kk = 0; kk < 72; ++kk) {
    // stage 32x256 bf16 weight slab; thread tid == co (64 B per thread)
    {
      const unsigned short* gsrc = WT + (size_t)tid * 2304 + kk * 32;
      unsigned short*       ldst = wslab + tid * 32;
#ifdef USE_ASYNC_LDS
      // gfx1250 async copy: immediate offset applies to both LDS and global addr
      __builtin_amdgcn_global_load_async_to_lds_b128((gv4i*)gsrc, (lv4i*)ldst, 0, 0);
      __builtin_amdgcn_global_load_async_to_lds_b128((gv4i*)gsrc, (lv4i*)ldst, 16, 0);
      __builtin_amdgcn_global_load_async_to_lds_b128((gv4i*)gsrc, (lv4i*)ldst, 32, 0);
      __builtin_amdgcn_global_load_async_to_lds_b128((gv4i*)gsrc, (lv4i*)ldst, 48, 0);
      __builtin_amdgcn_s_wait_asynccnt(0);
#else
      const uint4* src = (const uint4*)gsrc;
      uint4* dst = (uint4*)ldst;
      dst[0] = src[0]; dst[1] = src[1]; dst[2] = src[2]; dst[3] = src[3];
#endif
      if (kk + 1 < 72)
        __builtin_prefetch(WT + (size_t)tid * 2304 + (kk + 1) * 32, 0, 0);
    }
    __syncthreads();

    union { unsigned u[8]; v16bf v; } A0, A1, B0, B1;
#pragma unroll
    for (int p = 0; p < 8; ++p) {
      // A 16x32 bf16 layout: VGPR p holds K-pair (kb + {2p | 16+2(p-4)})
      int K0 = kb + ((p < 4) ? (2 * p) : (16 + 2 * (p - 4)));
      int kg = kk * 32 + K0;                   // global K (even)
      int ky = kg / 768;
      int rem = kg - ky * 768;
      int kx = rem >> 8;
      int ci = rem & 255;                      // even -> pair contiguous
      A0.u[p] = patchU[(((ky + 0) * 18 + m + kx) << 7) + (ci >> 1)];
      A1.u[p] = patchU[(((ky + 1) * 18 + m + kx) << 7) + (ci >> 1)];
    }
#pragma unroll
    for (int p = 0; p < 8; ++p) {
      // B 32x16 bf16 layout: lanes 0-15 K=0..7,16..23, lanes 16-31 K=8..15,24..31
      B0.u[p] = wslabU[co0 * 16 + khU + p + ((p < 4) ? 0 : 4)];
      B1.u[p] = wslabU[co1 * 16 + khU + p + ((p < 4) ? 0 : 4)];
    }
    acc00 = __builtin_amdgcn_wmma_f32_16x16x32_bf16(false, A0.v, false, B0.v,
                                                    (short)0, acc00, false, false);
    acc01 = __builtin_amdgcn_wmma_f32_16x16x32_bf16(false, A0.v, false, B1.v,
                                                    (short)0, acc01, false, false);
    acc10 = __builtin_amdgcn_wmma_f32_16x16x32_bf16(false, A1.v, false, B0.v,
                                                    (short)0, acc10, false, false);
    acc11 = __builtin_amdgcn_wmma_f32_16x16x32_bf16(false, A1.v, false, B1.v,
                                                    (short)0, acc11, false, false);
    __syncthreads();
  }

  // epilogue: bias + ReLU + bf16 store of t for both rows
  const int mhi = (lane < 16) ? 0 : 8;
  const size_t posBase = (size_t)n * TOTP + posOff + (size_t)y0 * W + x0;
  const float b0 = bconv[co0], b1 = bconv[co1];
#pragma unroll
  for (int r = 0; r < 8; ++r) {
    int M = r + mhi;
    size_t row0 = (posBase + M) * 256;
    size_t row1 = (posBase + W + M) * 256;
    T[row0 + co0] = f2bf(fmaxf(acc00[r] + b0, 0.0f));
    T[row0 + co1] = f2bf(fmaxf(acc01[r] + b1, 0.0f));
    T[row1 + co0] = f2bf(fmaxf(acc10[r] + b0, 0.0f));
    T[row1 + co1] = f2bf(fmaxf(acc11[r] + b1, 0.0f));
  }
}

// ---------------- 3: 1x1 heads + anchor decode ----------------
__global__ void __launch_bounds__(256) k_head(const unsigned short* __restrict__ T,
                                              const float* __restrict__ wobj,
                                              const float* __restrict__ bobj,
                                              const float* __restrict__ wdel,
                                              const float* __restrict__ bdel,
                                              float* __restrict__ SC,
                                              float* __restrict__ BX) {
  __shared__ float swo[256 * 3];
  __shared__ float swd[256 * 12];
  __shared__ float sbo[3], sbd[12];
  int tid = threadIdx.x;
  for (int i = tid; i < 768;  i += 256) swo[i] = wobj[i];
  for (int i = tid; i < 3072; i += 256) swd[i] = wdel[i];
  if (tid < 3)  sbo[tid] = bobj[tid];
  if (tid < 12) sbd[tid] = bdel[tid];
  __syncthreads();

  int q = blockIdx.x * 256 + tid;
  if (q >= NIMG * TOTP) return;
  int n  = q / TOTP;
  int pa = q - n * TOTP;
  int l = 0;
  if (pa >= d_prefP[1]) l = 1;
  if (pa >= d_prefP[2]) l = 2;
  if (pa >= d_prefP[3]) l = 3;
  if (pa >= d_prefP[4]) l = 4;
  int p = pa - d_prefP[l];
  int W = d_dim[l];

  float accs[3]  = {0.f, 0.f, 0.f};
  float accd[12] = {0.f,0.f,0.f,0.f,0.f,0.f,0.f,0.f,0.f,0.f,0.f,0.f};
  const unsigned* tp = (const unsigned*)(T + (size_t)q * 256);
  for (int c2 = 0; c2 < 128; ++c2) {
    unsigned u = tp[c2];
    float v0 = bf2f((unsigned short)(u & 0xFFFFu));
    float v1 = bf2f((unsigned short)(u >> 16));
    int c0 = c2 * 2, c1 = c0 + 1;
#pragma unroll
    for (int a = 0; a < 3; ++a)
      accs[a] += v0 * swo[c0 * 3 + a] + v1 * swo[c1 * 3 + a];
#pragma unroll
    for (int j = 0; j < 12; ++j)
      accd[j] += v0 * swd[c0 * 12 + j] + v1 * swd[c1 * 12 + j];
  }

  int yq = p / W, xq = p - yq * W;
  float stride = d_stride[l], sz = d_size[l];
  size_t sbase = (size_t)n * TOTP3 + d_off3[l] + (size_t)p * 3;
#pragma unroll
  for (int a = 0; a < 3; ++a) {
    SC[sbase + a] = accs[a] + sbo[a];
    float ratio = d_ratio[a];
    float wa = sqrtf(sz * sz / ratio);
    float ha = ratio * wa;
    float cxa = xq * stride, cya = yq * stride;
    float dx = accd[a * 4 + 0] + sbd[a * 4 + 0];
    float dy = accd[a * 4 + 1] + sbd[a * 4 + 1];
    float dw = fminf(accd[a * 4 + 2] + sbd[a * 4 + 2], CLIPV);
    float dh = fminf(accd[a * 4 + 3] + sbd[a * 4 + 3], CLIPV);
    float cx = dx * wa + cxa, cy = dy * ha + cya;
    float hw = 0.5f * expf(dw) * wa;
    float hh = 0.5f * expf(dh) * ha;
    float* bp = BX + (sbase + a) * 4;
    bp[0] = fminf(fmaxf(cx - hw, 0.f), IMGSZ);
    bp[1] = fminf(fmaxf(cy - hh, 0.f), IMGSZ);
    bp[2] = fminf(fmaxf(cx + hw, 0.f), IMGSZ);
    bp[3] = fminf(fmaxf(cy + hh, 0.f), IMGSZ);
  }
}

// ---------------- 4: per (image, level) exact top-k via 4-round radix select ----------------
__global__ void __launch_bounds__(1024) k_topk(const float* __restrict__ SC,
                                               const float* __restrict__ BX,
                                               float* __restrict__ CSC,
                                               float* __restrict__ CBX,
                                               int* __restrict__ CLV,
                                               unsigned* __restrict__ CNT) {
  __shared__ unsigned hist[256];
  __shared__ unsigned sh_b, sh_cum;
  int l = blockIdx.x, n = blockIdx.y, tid = threadIdx.x;
  int cnt = 3 * d_P[l];
  int k   = d_ksel[l];
  const float* s = SC + (size_t)n * TOTP3 + d_off3[l];

  unsigned prefix = 0, selG = 0;
  int krem = k;
  for (int shift = 24; shift >= 0; shift -= 8) {
    for (int i = tid; i < 256; i += 1024) hist[i] = 0u;
    __syncthreads();
    unsigned mask = (shift == 24) ? 0u : (0xFFFFFFFFu << (shift + 8));
    for (int i = tid; i < cnt; i += 1024) {
      unsigned key = fkey(s[i]);
      if ((key & mask) == prefix) atomicAdd(&hist[(key >> shift) & 255u], 1u);
    }
    __syncthreads();
    if (tid == 0) {
      unsigned cum = 0; int b = 255;
      for (; b > 0; --b) {
        if (cum + hist[b] >= (unsigned)krem) break;
        cum += hist[b];
      }
      sh_b = (unsigned)b; sh_cum = cum;
    }
    __syncthreads();
    prefix |= (sh_b << shift);
    selG += sh_cum;
    krem -= (int)sh_cum;
    __syncthreads();
  }
  const unsigned Tkey = prefix;           // exact threshold key
  unsigned* cG = &CNT[(n * 5 + l) * 2];
  unsigned* cE = cG + 1;
  for (int i = tid; i < cnt; i += 1024) {
    unsigned key = fkey(s[i]);
    int slot = -1;
    if (key > Tkey) {
      slot = (int)atomicAdd(cG, 1u);
    } else if (key == Tkey) {
      unsigned e = atomicAdd(cE, 1u);
      if (e < (unsigned)krem) slot = (int)(selG + e);
    }
    if (slot >= 0) {
      int dst = n * KCOMB + d_cbase[l] + slot;
      CSC[dst] = s[i];
      CLV[dst] = l;
      size_t src = (size_t)n * TOTP3 + d_off3[l] + i;
#pragma unroll
      for (int c = 0; c < 4; ++c) CBX[(size_t)dst * 4 + c] = BX[src * 4 + c];
    }
  }
}

// ---------------- 5: per-image bitonic sort (desc score, asc idx) ----------------
__global__ void __launch_bounds__(1024) k_sort(const float* __restrict__ CSC,
                                               const float* __restrict__ CBX,
                                               const int* __restrict__ CLV,
                                               float* __restrict__ SSC,
                                               float* __restrict__ SBX,
                                               int* __restrict__ SLV) {
  __shared__ float sk[8192];
  __shared__ unsigned short si[8192];
  int n = blockIdx.x, tid = threadIdx.x;
  for (int i = tid; i < 8192; i += 1024) {
    sk[i] = (i < KCOMB) ? CSC[n * KCOMB + i] : neg_inf();
    si[i] = (unsigned short)i;
  }
  __syncthreads();
  for (int ks = 2; ks <= 8192; ks <<= 1) {
    for (int j = ks >> 1; j > 0; j >>= 1) {
      for (int i = tid; i < 8192; i += 1024) {
        int l2 = i ^ j;
        if (l2 > i) {
          bool dir = ((i & ks) == 0);          // true = descending block
          float a = sk[i], b = sk[l2];
          unsigned short ia = si[i], ib = si[l2];
          bool aAfterB = (a < b) || (a == b && ia > ib);
          if (dir ? aAfterB : !aAfterB) {
            sk[i] = b; sk[l2] = a; si[i] = ib; si[l2] = ia;
          }
        }
      }
      __syncthreads();
    }
  }
  for (int i = tid; i < KCOMB; i += 1024) {
    int id = si[i];
    SSC[n * KCOMB + i] = sk[i];
    SLV[n * KCOMB + i] = CLV[n * KCOMB + id];
#pragma unroll
    for (int c = 0; c < 4; ++c)
      SBX[((size_t)n * KCOMB + i) * 4 + c] = CBX[((size_t)n * KCOMB + id) * 4 + c];
  }
}

// ---------------- 6: NMS suppression bitmask ----------------
__global__ void __launch_bounds__(256) k_mask(const float* __restrict__ SBX,
                                              const int* __restrict__ SLV,
                                              unsigned* __restrict__ MASK) {
  int t = blockIdx.x * 256 + threadIdx.x;
  int n = blockIdx.y;
  if (t >= KCOMB * NWORDS) return;
  int i  = t / NWORDS;
  int wb = t - i * NWORDS;
  float offi = SLV[n * KCOMB + i] * (2.0f * IMGSZ + 2.0f);
  const float* bi = SBX + ((size_t)n * KCOMB + i) * 4;
  float x0 = bi[0] + offi, y0 = bi[1] + offi, x1 = bi[2] + offi, y1 = bi[3] + offi;
  float ai = (x1 - x0) * (y1 - y0);
  unsigned word = 0;
  for (int b = 0; b < 32; ++b) {
    int j = wb * 32 + b;
    if (j > i) {
      float offj = SLV[n * KCOMB + j] * (2.0f * IMGSZ + 2.0f);
      const float* bj = SBX + ((size_t)n * KCOMB + j) * 4;
      float u0 = bj[0] + offj, v0 = bj[1] + offj, u1 = bj[2] + offj, v1 = bj[3] + offj;
      float aj = (u1 - u0) * (v1 - v0);
      float ix0 = fmaxf(x0, u0), iy0 = fmaxf(y0, v0);
      float ix1 = fminf(x1, u1), iy1 = fminf(y1, v1);
      float inter = fmaxf(ix1 - ix0, 0.f) * fmaxf(iy1 - iy0, 0.f);
      float iou = inter / (ai + aj - inter + 1e-9f);
      if (iou > NMS_THR) word |= (1u << b);
    }
  }
  MASK[((size_t)n * KCOMB + i) * NWORDS + wb] = word;
}

// ---------------- 7: greedy sequential NMS reduction ----------------
__global__ void __launch_bounds__(256) k_nms(const unsigned* __restrict__ MASK,
                                             unsigned* __restrict__ KEEP) {
  __shared__ unsigned rem[NWORDS];
  int n = blockIdx.x, tid = threadIdx.x;
  if (tid < NWORDS) rem[tid] = 0u;
  __syncthreads();
  for (int i = 0; i < KCOMB; ++i) {
    bool alive = !((rem[i >> 5] >> (i & 31)) & 1u);
    __syncthreads();
    if (alive && tid < NWORDS)
      rem[tid] |= MASK[((size_t)n * KCOMB + i) * NWORDS + tid];
    __syncthreads();
  }
  if (tid < NWORDS) KEEP[n * NWORDS + tid] = ~rem[tid];
}

// ---------------- 8: final top_k(masked, 1000) emulation ----------------
__global__ void __launch_bounds__(256) k_final(const float* __restrict__ SSC,
                                               const float* __restrict__ SBX,
                                               const unsigned* __restrict__ KEEP,
                                               float* __restrict__ out) {
  __shared__ unsigned kw[NWORDS];
  __shared__ int wp[NWORDS + 1];
  int n = blockIdx.x, tid = threadIdx.x;
  if (tid < NWORDS) kw[tid] = KEEP[n * NWORDS + tid];
  __syncthreads();
  if (tid == 0) {
    int c = 0;
    for (int w = 0; w < NWORDS; ++w) { wp[w] = c; c += __popc(kw[w]); }
    wp[NWORDS] = c;
  }
  __syncthreads();
  int Kk = wp[NWORDS];
  for (int i = tid; i < KCOMB; i += 256) {
    unsigned w = kw[i >> 5];
    bool kept = (w >> (i & 31)) & 1u;
    int before = wp[i >> 5] + __popc(w & ((1u << (i & 31)) - 1u));
    int rank = kept ? before : (Kk + (i - before));
    if (rank < 1000) {
      float* dst = out + ((size_t)n * 1000 + rank) * 5;
#pragma unroll
      for (int c = 0; c < 4; ++c) dst[c] = SBX[((size_t)n * KCOMB + i) * 4 + c];
      dst[4] = kept ? SSC[n * KCOMB + i] : neg_inf();
    }
  }
}

// ---------------- host launcher ----------------
extern "C" void kernel_launch(void* const* d_in, const int* in_sizes, int n_in,
                              void* d_out, int out_size, void* d_ws, size_t ws_size,
                              hipStream_t stream) {
  (void)in_sizes; (void)n_in; (void)out_size; (void)ws_size;
  const float* feats[5] = {(const float*)d_in[0], (const float*)d_in[1],
                           (const float*)d_in[2], (const float*)d_in[3],
                           (const float*)d_in[4]};
  const float* w_conv = (const float*)d_in[5];
  const float* b_conv = (const float*)d_in[6];
  const float* w_obj  = (const float*)d_in[7];
  const float* b_obj  = (const float*)d_in[8];
  const float* w_del  = (const float*)d_in[9];
  const float* b_del  = (const float*)d_in[10];
  float* out = (float*)d_out;

  char* ws = (char*)d_ws;
  size_t o = 0;
  auto alloc = [&](size_t bytes) {
    size_t r = o;
    o += (bytes + 255) & ~(size_t)255;
    return r;
  };
  unsigned short* WT  = (unsigned short*)(ws + alloc((size_t)2304 * 256 * 2));
  unsigned short* T   = (unsigned short*)(ws + alloc((size_t)NIMG * TOTP * 256 * 2));
  float*    SC   = (float*)(ws + alloc((size_t)NIMG * TOTP3 * 4));
  float*    BX   = (float*)(ws + alloc((size_t)NIMG * TOTP3 * 16));
  float*    CSC  = (float*)(ws + alloc((size_t)NIMG * KCOMB * 4));
  float*    CBX  = (float*)(ws + alloc((size_t)NIMG * KCOMB * 16));
  int*      CLV  = (int*)(ws + alloc((size_t)NIMG * KCOMB * 4));
  float*    SSC  = (float*)(ws + alloc((size_t)NIMG * KCOMB * 4));
  float*    SBX  = (float*)(ws + alloc((size_t)NIMG * KCOMB * 16));
  int*      SLV  = (int*)(ws + alloc((size_t)NIMG * KCOMB * 4));
  unsigned* MASK = (unsigned*)(ws + alloc((size_t)NIMG * KCOMB * NWORDS * 4));
  unsigned* KEEP = (unsigned*)(ws + alloc((size_t)NIMG * NWORDS * 4));
  unsigned* CNT  = (unsigned*)(ws + alloc(256));

  static const int hP[5]    = {65536, 16384, 4096, 1024, 256};
  static const int hdim[5]  = {256, 128, 64, 32, 16};
  static const int hpref[5] = {0, 65536, 81920, 86016, 87040};

  k_init<<<1, 64, 0, stream>>>(CNT);
  k_wconv<<<2304, 256, 0, stream>>>(w_conv, WT);
  for (int l = 0; l < 5; ++l) {
    dim3 g(hP[l] / 32, NIMG);   // 2-row x 16-col tiles
    k_conv3<<<g, 256, 0, stream>>>(feats[l], WT, b_conv, T, hdim[l], hdim[l], hpref[l]);
  }
  k_head<<<(NIMG * TOTP + 255) / 256, 256, 0, stream>>>(T, w_obj, b_obj, w_del, b_del, SC, BX);
  k_topk<<<dim3(5, NIMG), 1024, 0, stream>>>(SC, BX, CSC, CBX, CLV, CNT);
  k_sort<<<NIMG, 1024, 0, stream>>>(CSC, CBX, CLV, SSC, SBX, SLV);
  k_mask<<<dim3((KCOMB * NWORDS + 255) / 256, NIMG), 256, 0, stream>>>(SBX, SLV, MASK);
  k_nms<<<NIMG, 256, 0, stream>>>(MASK, KEEP);
  k_final<<<NIMG, 256, 0, stream>>>(SSC, SBX, KEEP, out);
}
